// SpatioConvLayer_12481174962762
// MI455X (gfx1250) — matrix operand order repaired and checked by reading, not compiled, not run on hardware
//
#include <hip/hip_runtime.h>
#include <hip/hip_bf16.h>

typedef __attribute__((ext_vector_type(16))) __bf16 v16bf;
typedef __attribute__((ext_vector_type(8)))  __bf16 v8bf;
typedef __attribute__((ext_vector_type(8)))  float  v8f;

#define BB   16
#define CI   64
#define CO   64
#define TT   48
#define NN   1024
#define KSZ  3
#define MC   128          // m-chunk width
#define XP   72           // XsT pitch (64 i + 8 pad), 144B rows (8/16B aligned)
#define YP   136          // Ys  pitch (128 m + 8 pad), 272B rows (16B aligned)

#define LKSW_ELEMS (3*32*64*512)   // 3,145,728 bf16 = 6 MB
#define THSW_ELEMS (3*4*2*32*16)   // 12,288 bf16

static __device__ __forceinline__ float f4c(const float4& v, int r) {
    switch (r) { case 0: return v.x; case 1: return v.y; case 2: return v.z; default: return v.w; }
}

// ---------------------------------------------------------------------------
// Prep: Lk -> bf16 transposed+swizzled into per-lane WMMA B-fragment order;
// theta -> bf16 swizzled into per-lane WMMA A-fragment order.
// B-frag (16-bit, 32x16): lane holds col = lane&15, K = e + 16*(lane>>4)
// A-frag (16-bit, 16x32): lane holds row = lane&15, K = (e<8 ? e+8h : e+8+8h)
// ---------------------------------------------------------------------------
__global__ __launch_bounds__(256) void spatio_prep(
    const float* __restrict__ lk, const float* __restrict__ theta,
    __bf16* __restrict__ lksw, __bf16* __restrict__ thsw)
{
    int idx = blockIdx.x * 256 + threadIdx.x;
    if (idx < LKSW_ELEMS) {
        int e    = idx & 15;
        int lane = (idx >> 4)  & 31;
        int nt   = (idx >> 9)  & 63;
        int mt   = (idx >> 15) & 31;
        int k    =  idx >> 20;
        int m = mt * 32 + e + ((lane >> 4) << 4);
        int n = nt * 16 + (lane & 15);
        lksw[idx] = (__bf16)lk[(k * 1024 + n) * 1024 + m];   // LkT[m][n] fragment
    } else {
        int j = idx - LKSW_ELEMS;
        if (j < THSW_ELEMS) {
            int e    = j & 15;
            int lane = (j >> 4)  & 31;
            int ks   = (j >> 9)  & 1;
            int rb   = (j >> 10) & 3;
            int k    =  j >> 12;
            int h = lane >> 4;
            int o = rb * 16 + (lane & 15);
            int K = (e < 8) ? (e + 8 * h) : (e + 8 + 8 * h);
            int i = ks * 32 + K;
            thsw[j] = (__bf16)theta[(i * CO + o) * KSZ + k]; // A[o][i] fragment
        }
    }
}

// ---------------------------------------------------------------------------
// Fused main kernel: one block = one (b,t) pair x one 256-wide n-tile.
// ---------------------------------------------------------------------------
__global__ __launch_bounds__(256) void spatio_main(
    const float*  __restrict__ x,
    const __bf16* __restrict__ lksw,
    const __bf16* __restrict__ thsw,
    const float*  __restrict__ bias,
    float*        __restrict__ out)
{
    __shared__ __bf16 XsT[MC][XP];        // X chunk, transposed: XsT[m][i]
    __shared__ __bf16 Ys[KSZ][CO][YP];    // Y[k][o][m] for current chunk

    const int bt   = blockIdx.x;
    const int bb   = bt / TT;
    const int t    = bt % TT;
    const int n0   = blockIdx.y * 256;
    const int tid  = threadIdx.x;
    const int w    = tid >> 5;
    const int lane = tid & 31;
    const int lh   = lane >> 4;           // half-wave select
    const int l15  = lane & 15;

    v8f acc[4][2];
    #pragma unroll
    for (int rb = 0; rb < 4; ++rb)
        #pragma unroll
        for (int jc = 0; jc < 2; ++jc)
            acc[rb][jc] = (v8f){0.f,0.f,0.f,0.f,0.f,0.f,0.f,0.f};

    const float* xbase = x + ((size_t)bb * CI * TT + t) * NN;  // row i at +i*TT*NN

    const int  mloc  = w * 16 + l15;                            // stage-A column (mb = w)
    const unsigned ybase = (unsigned)(uintptr_t)&Ys[0][lh * 8][mloc];

    for (int mc = 0; mc < NN / MC; ++mc) {
        const int m0 = mc * MC;

        // ---- load X chunk (fp32 -> bf16, transposed into LDS) ----
        // Each thread: a 4-row x 4-col micro-tile -> packed ds_store_b64s.
        #pragma unroll
        for (int q = 0; q < 2; ++q) {
            int p  = q * 256 + tid;            // 0..511
            int i0 = (p >> 5) << 2;            // 0..60 step 4
            int mm = (p & 31) << 2;            // 0..124 step 4
            const float* src = xbase + (size_t)i0 * (TT * NN) + m0 + mm;
            float4 r0 = *(const float4*)(src);
            float4 r1 = *(const float4*)(src + TT * NN);
            float4 r2 = *(const float4*)(src + 2 * TT * NN);
            float4 r3 = *(const float4*)(src + 3 * TT * NN);
            if (mc + 1 < NN / MC) {            // prefetch next chunk (global_prefetch_b8)
                __builtin_prefetch(src + MC, 0, 1);
            }
            #pragma unroll
            for (int r = 0; r < 4; ++r) {
                float a0 = f4c(r0, r), a1 = f4c(r1, r), a2 = f4c(r2, r), a3 = f4c(r3, r);
                unsigned lo, hi;
                asm("v_cvt_pk_bf16_f32 %0, %1, %2" : "=v"(lo) : "v"(a0), "v"(a1));
                asm("v_cvt_pk_bf16_f32 %0, %1, %2" : "=v"(hi) : "v"(a2), "v"(a3));
                uint2 uv; uv.x = lo; uv.y = hi;
                *(uint2*)(&XsT[mm + r][i0]) = uv;   // 4 bf16: rows i0..i0+3
            }
        }
        __syncthreads();

        // ---- stage A: Y[k] = theta_k^T @ Xchunk ----
        // Wave w owns m-block mb = w for ALL 12 (k, rb) tiles: the two XsT
        // B-fragments are loaded ONCE and reused by all 24 WMMAs; packed
        // results are buffered in registers and stored in one batch so the
        // load->wmma chain never waits on DS stores.
        union { v16bf v; v8bf h[2]; } ub[2];
        #pragma unroll
        for (int ks = 0; ks < 2; ++ks) {
            const __bf16* bp = &XsT[mloc][ks * 32 + lh * 8];
            ub[ks].h[0] = *(const v8bf*)(bp);
            ub[ks].h[1] = *(const v8bf*)(bp + 16);
        }
        unsigned preg[12][4];
        #pragma unroll
        for (int tt = 0; tt < 12; ++tt) {
            int k  = tt >> 2;
            int rb = tt & 3;
            v8f c = (v8f){0.f,0.f,0.f,0.f,0.f,0.f,0.f,0.f};
            #pragma unroll
            for (int ks = 0; ks < 2; ++ks) {
                v16bf a = *(const v16bf*)(thsw + ((((k * 4 + rb) * 2 + ks) * 32 + lane) << 4));
                c = __builtin_amdgcn_wmma_f32_16x16x32_bf16(
                        false, a, false, ub[ks].v, (short)0, c, false, false);
            }
            asm("v_cvt_pk_bf16_f32 %0, %1, %2" : "=v"(preg[tt][0]) : "v"(c[0]), "v"(c[1]));
            asm("v_cvt_pk_bf16_f32 %0, %1, %2" : "=v"(preg[tt][1]) : "v"(c[2]), "v"(c[3]));
            asm("v_cvt_pk_bf16_f32 %0, %1, %2" : "=v"(preg[tt][2]) : "v"(c[4]), "v"(c[5]));
            asm("v_cvt_pk_bf16_f32 %0, %1, %2" : "=v"(preg[tt][3]) : "v"(c[6]), "v"(c[7]));
        }
        // batched Ys stores: one base register, compile-time tile offsets
        #pragma unroll
        for (int tt = 0; tt < 12; ++tt) {
            int k  = tt >> 2;
            int rb = tt & 3;
            unsigned addr = ybase + (unsigned)(k * 17408 + rb * 4352); // k*64*YP*2 + rb*16*YP*2
            asm volatile(
                "ds_store_b16 %0, %1\n\t"
                "ds_store_b16_d16_hi %0, %1 offset:272\n\t"
                "ds_store_b16 %0, %2 offset:544\n\t"
                "ds_store_b16_d16_hi %0, %2 offset:816\n\t"
                "ds_store_b16 %0, %3 offset:1088\n\t"
                "ds_store_b16_d16_hi %0, %3 offset:1360\n\t"
                "ds_store_b16 %0, %4 offset:1632\n\t"
                "ds_store_b16_d16_hi %0, %4 offset:1904"
                :: "v"(addr), "v"(preg[tt][0]), "v"(preg[tt][1]),
                   "v"(preg[tt][2]), "v"(preg[tt][3])
                : "memory");
        }
        __syncthreads();

        // ---- stage B: acc += Y @ LkT  (wave owns 2 n-cols x 4 o-row-blocks) --
        #pragma unroll
        for (int k = 0; k < KSZ; ++k) {
            #pragma unroll
            for (int kap = 0; kap < 4; ++kap) {
                int mtg = mc * 4 + kap;
                size_t base = ((size_t)((k * 32 + mtg) * 64 + (blockIdx.y * 16 + w * 2)) << 9)
                              + ((size_t)lane << 4);
                v16bf bf0 = *(const v16bf*)(lksw + base);
                v16bf bf1 = *(const v16bf*)(lksw + base + 512);
                #pragma unroll
                for (int rb = 0; rb < 4; ++rb) {
                    const __bf16* ap = &Ys[k][rb * 16 + l15][kap * 32 + lh * 16];
                    union { v16bf v; v8bf h[2]; } ua;
                    ua.h[0] = *(const v8bf*)(ap);
                    ua.h[1] = *(const v8bf*)(ap + 8);
                    acc[rb][0] = __builtin_amdgcn_wmma_f32_16x16x32_bf16(
                        false, ua.v, false, bf0, (short)0, acc[rb][0], false, false);
                    acc[rb][1] = __builtin_amdgcn_wmma_f32_16x16x32_bf16(
                        false, ua.v, false, bf1, (short)0, acc[rb][1], false, false);
                }
            }
        }
        __syncthreads();
    }

    // ---- epilogue: + bias + residual x, relu, store ----
    #pragma unroll
    for (int rb = 0; rb < 4; ++rb) {
        #pragma unroll
        for (int jc = 0; jc < 2; ++jc) {
            int n = n0 + (w * 2 + jc) * 16 + l15;
            #pragma unroll
            for (int v = 0; v < 8; ++v) {
                int o = rb * 16 + lh * 8 + v;
                size_t gi = (((size_t)bb * CO + o) * TT + t) * NN + n;
                float val = acc[rb][jc][v] + bias[o] + x[gi];
                out[gi] = fmaxf(val, 0.0f);
            }
        }
    }
}

extern "C" void kernel_launch(void* const* d_in, const int* in_sizes, int n_in,
                              void* d_out, int out_size, void* d_ws, size_t ws_size,
                              hipStream_t stream) {
    const float* x     = (const float*)d_in[0];   // [16,64,48,1024]
    const float* lk    = (const float*)d_in[1];   // [3,1024,1024]
    const float* theta = (const float*)d_in[2];   // [64,64,3]
    const float* bias  = (const float*)d_in[3];   // [1,64,1,1]
    float*       out   = (float*)d_out;

    __bf16* lksw = (__bf16*)d_ws;                         // 6 MB
    __bf16* thsw = (__bf16*)((char*)d_ws + (size_t)LKSW_ELEMS * 2);

    int total = LKSW_ELEMS + THSW_ELEMS;                  // 3,158,016
    spatio_prep<<<(total + 255) / 256, 256, 0, stream>>>(lk, theta, lksw, thsw);

    dim3 grid(BB * TT, NN / 256, 1);
    spatio_main<<<grid, 256, 0, stream>>>(x, lksw, thsw, bias, out);
}
